// MolRGCN_2869038153789
// MI455X (gfx1250) — compile-verified, matrix operand order checked by reading, not run on hardware
//
#include <hip/hip_runtime.h>
#include <hip/hip_bf16.h>

#define N_NODES 100000
#define N_EDGES 1600000
#define R_REL   4
#define D_IN    38
#define HID     64
#define G_GRAPH 1000

typedef __attribute__((ext_vector_type(16))) _Float16 v16h;
typedef __attribute__((ext_vector_type(8)))  _Float16 v8h;
typedef __attribute__((ext_vector_type(8)))  float    v8f;

// ---------------------------------------------------------------- utilities
__global__ void zero_kernel(float4* __restrict__ p, long long n4) {
  long long i = (long long)blockIdx.x * blockDim.x + threadIdx.x;
  if (i < n4) p[i] = make_float4(0.f, 0.f, 0.f, 0.f);
}

// ------------------------------------------------- weight fragment builder
// Fragment layout: idx = ((s*4 + t)*32 + lane)*16 + j  (f16)
// k(s,lane,j) = s*32 + (lane>>4)*8 + (j<8 ? j : 8+j) ; o = t*16 + (lane&15)
// k <  Kconv : W[r=k/F][i=k%F][o]   (per-relation block)
// k <  Ktot  : root[k-Kconv][o]
// else       : 0
__global__ void build_frag(_Float16* __restrict__ dst, int nk, int F,
                           int Kconv, int Ktot,
                           const float* __restrict__ W,
                           const float* __restrict__ root) {
  int total = nk * 4 * 32 * 16;
  for (int idx = blockIdx.x * blockDim.x + threadIdx.x; idx < total;
       idx += gridDim.x * blockDim.x) {
    int j    = idx & 15;
    int lane = (idx >> 4) & 31;
    int t    = (idx >> 9) & 3;
    int s    = idx >> 11;
    int k = s * 32 + ((lane >> 4) * 8) + (j < 8 ? j : 8 + j);
    int o = t * 16 + (lane & 15);
    float v = 0.f;
    if (k < Kconv) {
      int r = k / F, f = k - r * F;
      v = W[((size_t)r * F + f) * HID + o];
    } else if (k < Ktot) {
      v = root[(size_t)(k - Kconv) * HID + o];
    }
    dst[idx] = (_Float16)v;
  }
}

// ------------------------------------------------------------ count pass
__global__ void count_kernel(const int* __restrict__ ei,
                             const int* __restrict__ et,
                             float* __restrict__ cnt) {
  int e = blockIdx.x * blockDim.x + threadIdx.x;
  if (e < N_EDGES) {
    int d = ei[N_EDGES + e];
    int r = et[e];
    atomicAdd(&cnt[(size_t)d * R_REL + r], 1.0f);
  }
}

__global__ void rcnt_kernel(float* __restrict__ cnt, int n) {
  int i = blockIdx.x * blockDim.x + threadIdx.x;
  if (i < n) cnt[i] = 1.0f / fmaxf(cnt[i], 1.0f);
}

// ------------------------------------------------------------ edge scatter
// thread -> (edge, feature); 64 feature slots per edge (F<=64)
__global__ void scatter_kernel(const float* __restrict__ xin, int F,
                               const int* __restrict__ ei,
                               const int* __restrict__ et,
                               float* __restrict__ sums) {
  long long tid = (long long)blockIdx.x * blockDim.x + threadIdx.x;
  int f = (int)(tid & 63);
  long long e = tid >> 6;
  if (e >= N_EDGES || f >= F) return;
  int s = ei[e];
  int d = ei[N_EDGES + e];
  int r = et[e];
  float v = xin[(size_t)s * F + f];
  atomicAdd(&sums[((size_t)d * R_REL + r) * F + f], v);
}

// ------------------------------------------- build f16 A-feature matrix
// Afeat[n][k]: k<4F -> mean(r=k/F, f=k%F); k<5F -> xroot; else 0
__global__ void normalize_kernel(const float* __restrict__ sums,
                                 const float* __restrict__ rcnt,
                                 const float* __restrict__ xroot,
                                 int F, int KP,
                                 _Float16* __restrict__ Afeat) {
  int n = blockIdx.x;
  int K4 = 4 * F;
  for (int k = threadIdx.x; k < KP; k += blockDim.x) {
    float v = 0.f;
    if (k < K4) {
      int r = k / F, f = k - r * F;
      v = sums[((size_t)n * R_REL + r) * F + f] * rcnt[(size_t)n * R_REL + r];
    } else if (k < K4 + F) {
      v = xroot[(size_t)n * F + (k - K4)];
    }
    Afeat[(size_t)n * KP + k] = (_Float16)v;
  }
}

// ------------------------------------------------------- dense WMMA layer
__device__ inline v16h load_afrag(const _Float16* __restrict__ Arow, int k0) {
  v8h lo = *(const v8h*)(Arow + k0);
  v8h hi = *(const v8h*)(Arow + k0 + 16);
  v16h a;
#pragma unroll
  for (int i = 0; i < 8; ++i) { a[i] = lo[i]; a[i + 8] = hi[i]; }
  return a;
}

// one wave per 16-node tile; out[16,64] via v_wmma_f32_16x16x32_f16,
// fused bias+ReLU+residual+LayerNorm epilogue.
__global__ void __launch_bounds__(128) dense_kernel(
    const _Float16* __restrict__ Afeat, int KP, int nk,
    const _Float16* __restrict__ Wfrag, const float* __restrict__ bias,
    int res_mode, const _Float16* __restrict__ WfragR,
    const float* __restrict__ res_bias, int res_k0,
    const float* __restrict__ x_prev,
    const float* __restrict__ ln_g, const float* __restrict__ ln_b,
    float* __restrict__ out, int ntiles) {
  const int lane = threadIdx.x & 31;
  const int wave = threadIdx.x >> 5;
  const int tile = blockIdx.x * 4 + wave;
  if (tile >= ntiles) return;          // wave-uniform: EXEC all-ones for WMMA
  const int m  = lane & 15;
  const int kh = lane >> 4;
  const _Float16* Arow = Afeat + (size_t)(tile * 16 + m) * KP;

  v8f acc[4] = {};
  for (int s = 0; s < nk; ++s) {
    const v16h a = load_afrag(Arow, s * 32 + kh * 8);
    const _Float16* wp = Wfrag + (size_t)s * 2048 + lane * 16;
#pragma unroll
    for (int t = 0; t < 4; ++t) {
      const v16h b = *(const v16h*)(wp + t * 512);
      acc[t] = __builtin_amdgcn_wmma_f32_16x16x32_f16(
          false, a, false, b, (short)0, acc[t], false, false);
    }
  }

  float bcol[4], gcol[4], bln[4];
#pragma unroll
  for (int t = 0; t < 4; ++t) {
    bcol[t] = bias[t * 16 + m];
    gcol[t] = ln_g[t * 16 + m];
    bln[t]  = ln_b[t * 16 + m];
  }

  float val[4][8];
#pragma unroll
  for (int t = 0; t < 4; ++t)
#pragma unroll
    for (int r = 0; r < 8; ++r)
      val[t][r] = fmaxf(acc[t][r] + bcol[t], 0.0f);   // relu(conv + b)

  if (res_mode) {                       // layer 0: += x @ res_w + res_b
    v8f rac[4] = {};
    for (int s = 0; s < 2; ++s) {
      const v16h a = load_afrag(Arow, res_k0 + s * 32 + kh * 8);
      const _Float16* wp = WfragR + (size_t)s * 2048 + lane * 16;
#pragma unroll
      for (int t = 0; t < 4; ++t) {
        const v16h b = *(const v16h*)(wp + t * 512);
        rac[t] = __builtin_amdgcn_wmma_f32_16x16x32_f16(
            false, a, false, b, (short)0, rac[t], false, false);
      }
    }
#pragma unroll
    for (int t = 0; t < 4; ++t) {
      float rb = res_bias[t * 16 + m];
#pragma unroll
      for (int r = 0; r < 8; ++r) val[t][r] += rac[t][r] + rb;
    }
  } else {                              // layers 1/2: += x_prev
#pragma unroll
    for (int r = 0; r < 8; ++r) {
      const float* xp = x_prev + (size_t)(tile * 16 + r + kh * 8) * HID + m;
      val[0][r] += xp[0];
      val[1][r] += xp[16];
      val[2][r] += xp[32];
      val[3][r] += xp[48];
    }
  }

  // LayerNorm over the 64 features of each row; a row lives in one 16-lane
  // half-wave, so xor-reduce with masks 1,2,4,8 only.
#pragma unroll
  for (int r = 0; r < 8; ++r) {
    float s1 = val[0][r] + val[1][r] + val[2][r] + val[3][r];
    s1 += __shfl_xor(s1, 1, 32);
    s1 += __shfl_xor(s1, 2, 32);
    s1 += __shfl_xor(s1, 4, 32);
    s1 += __shfl_xor(s1, 8, 32);
    const float mu = s1 * (1.0f / 64.0f);
    float d[4], s2 = 0.f;
#pragma unroll
    for (int t = 0; t < 4; ++t) { d[t] = val[t][r] - mu; s2 += d[t] * d[t]; }
    s2 += __shfl_xor(s2, 1, 32);
    s2 += __shfl_xor(s2, 2, 32);
    s2 += __shfl_xor(s2, 4, 32);
    s2 += __shfl_xor(s2, 8, 32);
    const float inv = rsqrtf(s2 * (1.0f / 64.0f) + 1e-5f);
    float* op = out + (size_t)(tile * 16 + r + kh * 8) * HID + m;
    op[0]  = d[0] * inv * gcol[0] + bln[0];
    op[16] = d[1] * inv * gcol[1] + bln[1];
    op[32] = d[2] * inv * gcol[2] + bln[2];
    op[48] = d[3] * inv * gcol[3] + bln[3];
  }
}

// --------------------------------------------------------------- pooling
__global__ void gcnt_kernel(const int* __restrict__ batch,
                            float* __restrict__ gcnt) {
  int n = blockIdx.x * blockDim.x + threadIdx.x;
  if (n < N_NODES) atomicAdd(&gcnt[batch[n]], 1.0f);
}

__global__ void pool_kernel(const float* __restrict__ x3,
                            const int* __restrict__ batch,
                            float* __restrict__ pooled) {
  long long tid = (long long)blockIdx.x * blockDim.x + threadIdx.x;
  int f = (int)(tid & 63);
  long long n = tid >> 6;
  if (n >= N_NODES) return;
  int g = batch[n];
  atomicAdd(&pooled[(size_t)g * HID + f], x3[(size_t)n * HID + f]);
}

// --------------------------------------------------------------- readout
__global__ void readout_kernel(const float* __restrict__ pooled,
                               const float* __restrict__ gcnt,
                               const float* __restrict__ w1,
                               const float* __restrict__ b1,
                               const float* __restrict__ w2,
                               const float* __restrict__ b2,
                               float* __restrict__ out) {
  __shared__ float sp[64];
  __shared__ float sh[64];
  int g = blockIdx.x, j = threadIdx.x;
  float rg = 1.0f / fmaxf(gcnt[g], 1.0f);
  sp[j] = pooled[(size_t)g * HID + j] * rg;
  __syncthreads();
  float acc = b1[j];
  for (int i = 0; i < 64; ++i) acc += sp[i] * w1[i * HID + j];
  sh[j] = fmaxf(acc, 0.0f) * w2[j];
  __syncthreads();
  if (j == 0) {
    float s = b2[0];
    for (int i = 0; i < 64; ++i) s += sh[i];
    out[g] = s;
  }
}

// ================================================================ launch
extern "C" void kernel_launch(void* const* d_in, const int* in_sizes, int n_in,
                              void* d_out, int out_size, void* d_ws,
                              size_t ws_size, hipStream_t stream) {
  (void)in_sizes; (void)n_in; (void)out_size; (void)ws_size;
  const float* x     = (const float*)d_in[0];
  const int*   ei    = (const int*)d_in[1];
  const int*   et    = (const int*)d_in[2];
  const int*   batch = (const int*)d_in[3];
  const float* W0    = (const float*)d_in[4];
  const float* root0 = (const float*)d_in[5];
  const float* b0    = (const float*)d_in[6];
  const float* ln0g  = (const float*)d_in[7];
  const float* ln0b  = (const float*)d_in[8];
  const float* resw  = (const float*)d_in[9];
  const float* resb  = (const float*)d_in[10];
  const float* W1    = (const float*)d_in[11];
  const float* root1 = (const float*)d_in[12];
  const float* b1    = (const float*)d_in[13];
  const float* ln1g  = (const float*)d_in[14];
  const float* ln1b  = (const float*)d_in[15];
  const float* W2    = (const float*)d_in[16];
  const float* root2 = (const float*)d_in[17];
  const float* b2    = (const float*)d_in[18];
  const float* ln2g  = (const float*)d_in[19];
  const float* ln2b  = (const float*)d_in[20];
  const float* row1  = (const float*)d_in[21];
  const float* rob1  = (const float*)d_in[22];
  const float* row2  = (const float*)d_in[23];
  const float* rob2  = (const float*)d_in[24];

  // ---- workspace carve (256B aligned)
  char* p = (char*)d_ws;
  auto carve = [&](size_t bytes) -> void* {
    void* r = (void*)p;
    p += (bytes + 255) & ~(size_t)255;
    return r;
  };
  float*     sums   = (float*)carve((size_t)N_NODES * R_REL * HID * 4);   // 102.4 MB
  float*     rcnt   = (float*)carve((size_t)N_NODES * R_REL * 4);
  _Float16*  Afeat  = (_Float16*)carve((size_t)N_NODES * 320 * 2);        // 64 MB
  float*     xb0    = (float*)carve((size_t)N_NODES * HID * 4);
  float*     xb1    = (float*)carve((size_t)N_NODES * HID * 4);
  _Float16*  frag0  = (_Float16*)carve((size_t)6  * 2048 * 2);
  _Float16*  fragR  = (_Float16*)carve((size_t)2  * 2048 * 2);
  _Float16*  frag1  = (_Float16*)carve((size_t)10 * 2048 * 2);
  _Float16*  frag2  = (_Float16*)carve((size_t)10 * 2048 * 2);
  float*     pooled = (float*)carve((size_t)G_GRAPH * HID * 4);
  float*     gcnt   = (float*)carve((size_t)G_GRAPH * 4);

  const int ntiles = N_NODES / 16;                 // 6250 exact
  const int dense_blocks = (ntiles + 3) / 4;

  // ---- weight fragment prep (tiny)
  build_frag<<<48, 256, 0, stream>>>(frag0, 6, D_IN, 4 * D_IN, 5 * D_IN, W0, root0);
  build_frag<<<16, 256, 0, stream>>>(fragR, 2, D_IN, 0, D_IN, nullptr, resw);
  build_frag<<<80, 256, 0, stream>>>(frag1, 10, HID, 4 * HID, 5 * HID, W1, root1);
  build_frag<<<80, 256, 0, stream>>>(frag2, 10, HID, 4 * HID, 5 * HID, W2, root2);

  // ---- per-(node,relation) reciprocal counts (shared by all layers)
  {
    long long n4 = (long long)N_NODES * R_REL / 4;
    zero_kernel<<<(unsigned)((n4 + 255) / 256), 256, 0, stream>>>((float4*)rcnt, n4);
  }
  count_kernel<<<(N_EDGES + 255) / 256, 256, 0, stream>>>(ei, et, rcnt);
  rcnt_kernel<<<(N_NODES * R_REL + 255) / 256, 256, 0, stream>>>(rcnt, N_NODES * R_REL);

  // ================= layer 0 (D_IN -> H, GEMM residual) =================
  {
    long long n4 = (long long)N_NODES * R_REL * D_IN / 4;
    zero_kernel<<<(unsigned)((n4 + 255) / 256), 256, 0, stream>>>((float4*)sums, n4);
    long long nth = (long long)N_EDGES * 64;
    scatter_kernel<<<(unsigned)(nth / 256), 256, 0, stream>>>(x, D_IN, ei, et, sums);
    normalize_kernel<<<N_NODES, 256, 0, stream>>>(sums, rcnt, x, D_IN, 224, Afeat);
    dense_kernel<<<dense_blocks, 128, 0, stream>>>(
        Afeat, 224, 6, frag0, b0, /*res_mode=*/1, fragR, resb, /*res_k0=*/152,
        nullptr, ln0g, ln0b, xb0, ntiles);
  }
  // ================= layer 1 (H -> H, identity residual) ================
  {
    long long n4 = (long long)N_NODES * R_REL * HID / 4;
    zero_kernel<<<(unsigned)((n4 + 255) / 256), 256, 0, stream>>>((float4*)sums, n4);
    long long nth = (long long)N_EDGES * 64;
    scatter_kernel<<<(unsigned)(nth / 256), 256, 0, stream>>>(xb0, HID, ei, et, sums);
    normalize_kernel<<<N_NODES, 256, 0, stream>>>(sums, rcnt, xb0, HID, 320, Afeat);
    dense_kernel<<<dense_blocks, 128, 0, stream>>>(
        Afeat, 320, 10, frag1, b1, 0, nullptr, nullptr, 0,
        xb0, ln1g, ln1b, xb1, ntiles);
  }
  // ================= layer 2 =================
  {
    long long n4 = (long long)N_NODES * R_REL * HID / 4;
    zero_kernel<<<(unsigned)((n4 + 255) / 256), 256, 0, stream>>>((float4*)sums, n4);
    long long nth = (long long)N_EDGES * 64;
    scatter_kernel<<<(unsigned)(nth / 256), 256, 0, stream>>>(xb1, HID, ei, et, sums);
    normalize_kernel<<<N_NODES, 256, 0, stream>>>(sums, rcnt, xb1, HID, 320, Afeat);
    dense_kernel<<<dense_blocks, 128, 0, stream>>>(
        Afeat, 320, 10, frag2, b2, 0, nullptr, nullptr, 0,
        xb1, ln2g, ln2b, xb0, ntiles);
  }

  // ================= pooling + readout =================
  {
    long long n4 = (long long)G_GRAPH * HID / 4;
    zero_kernel<<<(unsigned)((n4 + 255) / 256), 256, 0, stream>>>((float4*)pooled, n4);
    long long g4 = G_GRAPH / 4;
    zero_kernel<<<(unsigned)((g4 + 255) / 256), 256, 0, stream>>>((float4*)gcnt, g4);
  }
  gcnt_kernel<<<(N_NODES + 255) / 256, 256, 0, stream>>>(batch, gcnt);
  {
    long long nth = (long long)N_NODES * 64;
    pool_kernel<<<(unsigned)((nth + 255) / 256), 256, 0, stream>>>(xb0, batch, pooled);
  }
  readout_kernel<<<G_GRAPH, 64, 0, stream>>>(pooled, gcnt, row1, rob1, row2, rob2,
                                             (float*)d_out);
}
